// biLSTM_50809463112204
// MI455X (gfx1250) — compile-verified, matrix operand order checked by reading, not run on hardware
//
#include <hip/hip_runtime.h>
#include <hip/hip_bf16.h>
#include <stdint.h>

// Problem constants (from reference): T=4096, E=512, H=1024, O=64
#define T_STEPS 4096
#define E_DIM   512
#define H_DIM   1024
#define G_DIM   4096   // 4*H (gate dim)
#define C_DIM   2048   // 2*H (bidirectional concat)
#define O_DIM   64

typedef __attribute__((ext_vector_type(16))) __bf16 v16bf;
typedef __attribute__((ext_vector_type(8)))  float  v8f;

union FragAB {
  v16bf v;
  uint4 q[2];
};

#define WMMA_BF16(A_, B_, C_)                                             \
  __builtin_amdgcn_wmma_f32_16x16x32_bf16(false, (A_), false, (B_),        \
                                          (short)0, (C_), false, false)

// ---------------------------------------------------------------------------
// Elementwise packing kernels
// ---------------------------------------------------------------------------
__global__ void f32_to_bf16_kernel(const float* __restrict__ src,
                                   __hip_bfloat16* __restrict__ dst, int n) {
  int i = blockIdx.x * blockDim.x + threadIdx.x;
  if (i < n) dst[i] = __float2bfloat16(src[i]);
}

__global__ void bias_sum_kernel(const float* __restrict__ a,
                                const float* __restrict__ b,
                                float* __restrict__ out, int n) {
  int i = blockIdx.x * blockDim.x + threadIdx.x;
  if (i < n) out[i] = a[i] + b[i];
}

// ---------------------------------------------------------------------------
// WMMA bf16 GEMM:  C[dir] = A[M,K] * B[dir][N,K]^T + bias[dir]   (f32 accum)
// grid = (N/512, M/32, 2 dirs), block = 256 (8 waves).
// Each wave owns a 32x64 output strip: 2 A fragments x 4 B fragments ->
// 8 named accumulators, so each B fragment load feeds 2 WMMAs.
// The 32x32 A tile is double-buffered in LDS (1 barrier per K-step).
// ---------------------------------------------------------------------------
__global__ __launch_bounds__(256)
void gemm_bf16_wmma(const __hip_bfloat16* __restrict__ A,   // [M,K]
                    const __hip_bfloat16* __restrict__ B0,  // [N,K] dir 0
                    const __hip_bfloat16* __restrict__ B1,  // [N,K] dir 1
                    const float* __restrict__ bias0,        // [N]
                    const float* __restrict__ bias1,        // [N]
                    float* __restrict__ C0,                 // [M,N]
                    float* __restrict__ C1,                 // [M,N]
                    int M, int N, int K) {
  const __hip_bfloat16* B    = blockIdx.z ? B1 : B0;
  const float*          bias = blockIdx.z ? bias1 : bias0;
  float*                C    = blockIdx.z ? C1 : C0;

  // 2 buffers x 32 rows x 32 bf16 (16 u32 per row)
  __shared__ __align__(16) unsigned lds_a[2][32 * 16];

  const int tid  = threadIdx.x;
  const int wave = tid >> 5;
  const int lane = tid & 31;
  const int half = lane >> 4;   // 0: lanes 0-15, 1: lanes 16-31
  const int l16  = lane & 15;

  const int m0    = blockIdx.y * 32;
  const int nbase = (blockIdx.x * 8 + wave) * 64;

  // staging: each thread loads 2 u32 (rows arow and arow+16, same K pair)
  const int arow  = tid >> 4;   // 0..15
  const int apair = tid & 15;   // 0..15 (u32 = 2 bf16)
  const __hip_bfloat16* As0 = A + (size_t)(m0 + arow) * K + apair * 2;
  const __hip_bfloat16* As1 = A + (size_t)(m0 + arow + 16) * K + apair * 2;

  // per-wave B row pointers (one row per N-tile; lane picks row l16 of tile)
  const __hip_bfloat16* br0 = B + (size_t)(nbase + 0 * 16 + l16) * K;
  const __hip_bfloat16* br1 = B + (size_t)(nbase + 1 * 16 + l16) * K;
  const __hip_bfloat16* br2 = B + (size_t)(nbase + 2 * 16 + l16) * K;
  const __hip_bfloat16* br3 = B + (size_t)(nbase + 3 * 16 + l16) * K;

  v8f accA0 = {}, accA1 = {}, accA2 = {}, accA3 = {};
  v8f accB0 = {}, accB1 = {}, accB2 = {}, accB3 = {};

  // preload tile 0
  lds_a[0][arow * 16 + apair]        = *reinterpret_cast<const unsigned*>(As0);
  lds_a[0][(arow + 16) * 16 + apair] = *reinterpret_cast<const unsigned*>(As1);
  __syncthreads();

  int buf = 0;
  for (int k0 = 0; k0 < K; k0 += 32) {
    const bool more = (k0 + 32) < K;
    if (more) {
      // stage next A tile into the alternate buffer
      lds_a[buf ^ 1][arow * 16 + apair] =
          *reinterpret_cast<const unsigned*>(As0 + k0 + 32);
      lds_a[buf ^ 1][(arow + 16) * 16 + apair] =
          *reinterpret_cast<const unsigned*>(As1 + k0 + 32);
      // near-cache prefetch of the K-after-next B chunk
      __builtin_prefetch(br0 + k0 + 64, 0, 3);
      __builtin_prefetch(br1 + k0 + 64, 0, 3);
      __builtin_prefetch(br2 + k0 + 64, 0, 3);
      __builtin_prefetch(br3 + k0 + 64, 0, 3);
    }

    // A fragments per ISA 16-bit 16x32 layout:
    // lanes 0-15 : M=lane,   K = {0..7, 16..23}
    // lanes 16-31: M=lane-16,K = {8..15,24..31}
    FragAB a0, a1;
    {
      const uint4* aq0 = reinterpret_cast<const uint4*>(&lds_a[buf][l16 * 16]);
      const uint4* aq1 =
          reinterpret_cast<const uint4*>(&lds_a[buf][(16 + l16) * 16]);
      a0.q[0] = aq0[half];
      a0.q[1] = aq0[2 + half];
      a1.q[0] = aq1[half];
      a1.q[1] = aq1[2 + half];
    }

    FragAB b0, b1, b2, b3;
    {
      const uint4* q0 = reinterpret_cast<const uint4*>(br0 + k0);
      const uint4* q1 = reinterpret_cast<const uint4*>(br1 + k0);
      const uint4* q2 = reinterpret_cast<const uint4*>(br2 + k0);
      const uint4* q3 = reinterpret_cast<const uint4*>(br3 + k0);
      b0.q[0] = q0[half]; b0.q[1] = q0[2 + half];
      b1.q[0] = q1[half]; b1.q[1] = q1[2 + half];
      b2.q[0] = q2[half]; b2.q[1] = q2[2 + half];
      b3.q[0] = q3[half]; b3.q[1] = q3[2 + half];
    }

    accA0 = WMMA_BF16(a0.v, b0.v, accA0);
    accB0 = WMMA_BF16(a1.v, b0.v, accB0);
    accA1 = WMMA_BF16(a0.v, b1.v, accA1);
    accB1 = WMMA_BF16(a1.v, b1.v, accB1);
    accA2 = WMMA_BF16(a0.v, b2.v, accA2);
    accB2 = WMMA_BF16(a1.v, b2.v, accB2);
    accA3 = WMMA_BF16(a0.v, b3.v, accA3);
    accB3 = WMMA_BF16(a1.v, b3.v, accB3);

    __syncthreads();  // staging done + all LDS reads done for this step
    buf ^= 1;
  }

  // C layout: VGPR r -> (M = r + 8*half, N = l16) within each 16x16 tile
  const v8f accTop[4] = {accA0, accA1, accA2, accA3};  // rows m0 .. m0+15
  const v8f accBot[4] = {accB0, accB1, accB2, accB3};  // rows m0+16 .. m0+31
#pragma unroll
  for (int j = 0; j < 4; ++j) {
    const int col  = nbase + j * 16 + l16;
    const float bv = bias[col];
#pragma unroll
    for (int r = 0; r < 8; ++r) {
      const int row = m0 + r + half * 8;
      C[(size_t)row * N + col]        = accTop[j][r] + bv;
      C[(size_t)(row + 16) * N + col] = accBot[j][r] + bv;
    }
  }
}

// ---------------------------------------------------------------------------
// Persistent recurrent LSTM kernel. grid = 2 blocks (dir 0 = fwd, 1 = bwd),
// block = 1024 threads. h lives in LDS, c in registers, w_hh (bf16) streams
// from L2 (8 MB/step/dir). Thread j owns output unit j (4 gate rows).
// ---------------------------------------------------------------------------
__device__ inline float mac8(float acc, uint4 q, float4 h0, float4 h1) {
  acc = fmaf(__uint_as_float(q.x << 16),          h0.x, acc);
  acc = fmaf(__uint_as_float(q.x & 0xffff0000u),  h0.y, acc);
  acc = fmaf(__uint_as_float(q.y << 16),          h0.z, acc);
  acc = fmaf(__uint_as_float(q.y & 0xffff0000u),  h0.w, acc);
  acc = fmaf(__uint_as_float(q.z << 16),          h1.x, acc);
  acc = fmaf(__uint_as_float(q.z & 0xffff0000u),  h1.y, acc);
  acc = fmaf(__uint_as_float(q.w << 16),          h1.z, acc);
  acc = fmaf(__uint_as_float(q.w & 0xffff0000u),  h1.w, acc);
  return acc;
}

__device__ inline float sigm(float x) { return 1.f / (1.f + __expf(-x)); }

__global__ __launch_bounds__(1024)
void lstm_recurrent(const float* __restrict__ pre_f,   // [T, 4H] natural t order
                    const float* __restrict__ pre_b,   // [T, 4H] natural t order
                    const __hip_bfloat16* __restrict__ whh_f,  // [4H, H]
                    const __hip_bfloat16* __restrict__ whh_b,  // [4H, H]
                    float* __restrict__ outF,          // [T, 2H]
                    __hip_bfloat16* __restrict__ outBF // [T, 2H]
                    ) {
  const int dir = blockIdx.x;  // 0 = forward, 1 = backward
  const float* pre          = dir ? pre_b : pre_f;
  const __hip_bfloat16* whh = dir ? whh_b : whh_f;
  const int j = threadIdx.x;   // 0..1023

  __shared__ __align__(16) float sh_h[H_DIM];
  sh_h[j] = 0.f;
  float c_reg = 0.f;
  __syncthreads();

  const uint4* w0 = reinterpret_cast<const uint4*>(whh + (size_t)(0 * H_DIM + j) * H_DIM);
  const uint4* w1 = reinterpret_cast<const uint4*>(whh + (size_t)(1 * H_DIM + j) * H_DIM);
  const uint4* w2 = reinterpret_cast<const uint4*>(whh + (size_t)(2 * H_DIM + j) * H_DIM);
  const uint4* w3 = reinterpret_cast<const uint4*>(whh + (size_t)(3 * H_DIM + j) * H_DIM);

  for (int s = 0; s < T_STEPS; ++s) {
    const int t = dir ? (T_STEPS - 1 - s) : s;

    float a0 = 0.f, a1 = 0.f, a2 = 0.f, a3 = 0.f;
    const float4* hv = reinterpret_cast<const float4*>(sh_h);
#pragma unroll 4
    for (int kk = 0; kk < H_DIM / 8; ++kk) {
      const float4 h0 = hv[kk * 2 + 0];
      const float4 h1 = hv[kk * 2 + 1];
      a0 = mac8(a0, w0[kk], h0, h1);
      a1 = mac8(a1, w1[kk], h0, h1);
      a2 = mac8(a2, w2[kk], h0, h1);
      a3 = mac8(a3, w3[kk], h0, h1);
    }

    const float* prow = pre + (size_t)t * G_DIM;
    const float gi = prow[0 * H_DIM + j] + a0;   // gate order: i, f, g, o
    const float gf = prow[1 * H_DIM + j] + a1;
    const float gg = prow[2 * H_DIM + j] + a2;
    const float go = prow[3 * H_DIM + j] + a3;

    c_reg = sigm(gf) * c_reg + sigm(gi) * tanhf(gg);
    const float hnew = sigm(go) * tanhf(c_reg);

    __syncthreads();                 // all reads of sh_h complete
    sh_h[j] = hnew;
    const size_t oidx = (size_t)t * C_DIM + (size_t)dir * H_DIM + j;
    outF[oidx]  = hnew;
    outBF[oidx] = __float2bfloat16(hnew);
    __syncthreads();                 // new h visible before next step
  }
}

// ---------------------------------------------------------------------------
// Final head: logits = c2 @ lin_w^T + lin_b, softmax over O=64.
// One 64-thread block per timestep.
// ---------------------------------------------------------------------------
__global__ __launch_bounds__(64)
void linear_softmax(const float* __restrict__ c2,   // [T, 2H]
                    const float* __restrict__ w,    // [O, 2H]
                    const float* __restrict__ b,    // [O]
                    float* __restrict__ out) {      // [T, 1, O]
  const int t = blockIdx.x;
  const int o = threadIdx.x;

  const float* xrow = c2 + (size_t)t * C_DIM;
  const float* wrow = w + (size_t)o * C_DIM;
  float acc = b[o];
  for (int k = 0; k < C_DIM; k += 4) {
    const float4 xv = *reinterpret_cast<const float4*>(xrow + k);
    const float4 wv = *reinterpret_cast<const float4*>(wrow + k);
    acc = fmaf(xv.x, wv.x, acc);
    acc = fmaf(xv.y, wv.y, acc);
    acc = fmaf(xv.z, wv.z, acc);
    acc = fmaf(xv.w, wv.w, acc);
  }

  __shared__ float red[O_DIM];
  red[o] = acc;
  __syncthreads();
  float m = red[0];
#pragma unroll
  for (int i = 1; i < O_DIM; ++i) m = fmaxf(m, red[i]);
  __syncthreads();
  const float e = __expf(acc - m);
  red[o] = e;
  __syncthreads();
  float sum = 0.f;
#pragma unroll
  for (int i = 0; i < O_DIM; ++i) sum += red[i];
  out[(size_t)t * O_DIM + o] = e / sum;
}

// ---------------------------------------------------------------------------
// Host orchestration
// ---------------------------------------------------------------------------
extern "C" void kernel_launch(void* const* d_in, const int* in_sizes, int n_in,
                              void* d_out, int out_size, void* d_ws, size_t ws_size,
                              hipStream_t stream) {
  const float* x      = (const float*)d_in[0];
  const float* wih_f1 = (const float*)d_in[1];
  const float* whh_f1 = (const float*)d_in[2];
  const float* bih_f1 = (const float*)d_in[3];
  const float* bhh_f1 = (const float*)d_in[4];
  const float* wih_b1 = (const float*)d_in[5];
  const float* whh_b1 = (const float*)d_in[6];
  const float* bih_b1 = (const float*)d_in[7];
  const float* bhh_b1 = (const float*)d_in[8];
  const float* wih_f2 = (const float*)d_in[9];
  const float* whh_f2 = (const float*)d_in[10];
  const float* bih_f2 = (const float*)d_in[11];
  const float* bhh_f2 = (const float*)d_in[12];
  const float* wih_b2 = (const float*)d_in[13];
  const float* whh_b2 = (const float*)d_in[14];
  const float* bih_b2 = (const float*)d_in[15];
  const float* bhh_b2 = (const float*)d_in[16];
  const float* lin_w  = (const float*)d_in[17];
  const float* lin_b  = (const float*)d_in[18];

  char* ws = (char*)d_ws;
  size_t off = 0;
  auto alloc = [&](size_t bytes) -> char* {
    char* p = ws + off;
    off += (bytes + 255) & ~(size_t)255;
    return p;
  };

  const size_t n_xs   = (size_t)T_STEPS * E_DIM;
  const size_t n_wih1 = (size_t)G_DIM * E_DIM;
  const size_t n_whh  = (size_t)G_DIM * H_DIM;
  const size_t n_wih2 = (size_t)G_DIM * C_DIM;

  __hip_bfloat16* xs_bf     = (__hip_bfloat16*)alloc(n_xs * 2);
  __hip_bfloat16* wihf1_bf  = (__hip_bfloat16*)alloc(n_wih1 * 2);
  __hip_bfloat16* wihb1_bf  = (__hip_bfloat16*)alloc(n_wih1 * 2);
  __hip_bfloat16* whhf1_bf  = (__hip_bfloat16*)alloc(n_whh * 2);
  __hip_bfloat16* whhb1_bf  = (__hip_bfloat16*)alloc(n_whh * 2);
  __hip_bfloat16* whhf2_bf  = (__hip_bfloat16*)alloc(n_whh * 2);
  __hip_bfloat16* whhb2_bf  = (__hip_bfloat16*)alloc(n_whh * 2);
  __hip_bfloat16* wihf2_bf  = (__hip_bfloat16*)alloc(n_wih2 * 2);
  __hip_bfloat16* wihb2_bf  = (__hip_bfloat16*)alloc(n_wih2 * 2);
  float* bias_f1 = (float*)alloc(G_DIM * 4);
  float* bias_b1 = (float*)alloc(G_DIM * 4);
  float* bias_f2 = (float*)alloc(G_DIM * 4);
  float* bias_b2 = (float*)alloc(G_DIM * 4);
  float* pre_a   = (float*)alloc((size_t)T_STEPS * G_DIM * 4);  // reused L1->L2
  float* pre_b   = (float*)alloc((size_t)T_STEPS * G_DIM * 4);  // reused L1->L2
  float* c1_f32  = (float*)alloc((size_t)T_STEPS * C_DIM * 4);
  __hip_bfloat16* c1_bf = (__hip_bfloat16*)alloc((size_t)T_STEPS * C_DIM * 2);
  float* c2_f32  = (float*)alloc((size_t)T_STEPS * C_DIM * 4);
  __hip_bfloat16* c2_bf = (__hip_bfloat16*)alloc((size_t)T_STEPS * C_DIM * 2);
  (void)ws_size; (void)in_sizes; (void)n_in; (void)out_size;

  auto conv = [&](const float* src, __hip_bfloat16* dst, size_t n) {
    f32_to_bf16_kernel<<<(unsigned)((n + 255) / 256), 256, 0, stream>>>(src, dst, (int)n);
  };
  conv(x,      xs_bf,    n_xs);
  conv(wih_f1, wihf1_bf, n_wih1);
  conv(wih_b1, wihb1_bf, n_wih1);
  conv(whh_f1, whhf1_bf, n_whh);
  conv(whh_b1, whhb1_bf, n_whh);
  conv(whh_f2, whhf2_bf, n_whh);
  conv(whh_b2, whhb2_bf, n_whh);
  conv(wih_f2, wihf2_bf, n_wih2);
  conv(wih_b2, wihb2_bf, n_wih2);

  bias_sum_kernel<<<G_DIM / 256, 256, 0, stream>>>(bih_f1, bhh_f1, bias_f1, G_DIM);
  bias_sum_kernel<<<G_DIM / 256, 256, 0, stream>>>(bih_b1, bhh_b1, bias_b1, G_DIM);
  bias_sum_kernel<<<G_DIM / 256, 256, 0, stream>>>(bih_f2, bhh_f2, bias_f2, G_DIM);
  bias_sum_kernel<<<G_DIM / 256, 256, 0, stream>>>(bih_b2, bhh_b2, bias_b2, G_DIM);

  // Layer 1 input projections (both directions via grid.z)
  {
    dim3 grid(G_DIM / 512, T_STEPS / 32, 2), block(256);
    gemm_bf16_wmma<<<grid, block, 0, stream>>>(
        xs_bf, wihf1_bf, wihb1_bf, bias_f1, bias_b1, pre_a, pre_b,
        T_STEPS, G_DIM, E_DIM);
  }
  // Layer 1 recurrence (fwd + bwd concurrently)
  lstm_recurrent<<<2, 1024, 0, stream>>>(pre_a, pre_b, whhf1_bf, whhb1_bf,
                                         c1_f32, c1_bf);
  // Layer 2 input projections (K = 2H = 2048), pre buffers reused
  {
    dim3 grid(G_DIM / 512, T_STEPS / 32, 2), block(256);
    gemm_bf16_wmma<<<grid, block, 0, stream>>>(
        c1_bf, wihf2_bf, wihb2_bf, bias_f2, bias_b2, pre_a, pre_b,
        T_STEPS, G_DIM, C_DIM);
  }
  // Layer 2 recurrence
  lstm_recurrent<<<2, 1024, 0, stream>>>(pre_a, pre_b, whhf2_bf, whhb2_bf,
                                         c2_f32, c2_bf);
  // Head
  linear_softmax<<<T_STEPS, 64, 0, stream>>>(c2_f32, lin_w, lin_b, (float*)d_out);
}
